// EdgeClassifier_14688788152882
// MI455X (gfx1250) — matrix operand order repaired and checked by reading, not compiled, not run on hardware
//
#include <hip/hip_runtime.h>
#include <hip/hip_bf16.h>

// ---------- types ----------
typedef __attribute__((ext_vector_type(16))) __bf16 bf16x16;
typedef __attribute__((ext_vector_type(8)))  __bf16 bf16x8;
typedef __attribute__((ext_vector_type(8)))  float  f32x8;

__device__ __forceinline__ __bf16 to_bf16(float f) {
  unsigned u = __builtin_bit_cast(unsigned, f);
  u += 0x7FFFu + ((u >> 16) & 1u);            // round-to-nearest-even
  unsigned short h = (unsigned short)(u >> 16);
  return __builtin_bit_cast(__bf16, h);
}

__device__ __forceinline__ bf16x16 cat8(bf16x8 lo, bf16x8 hi) {
  return __builtin_shufflevector(lo, hi, 0,1,2,3,4,5,6,7,8,9,10,11,12,13,14,15);
}

__device__ __forceinline__ f32x8 wmma_bf16(bf16x16 a, bf16x16 b, f32x8 c) {
  return __builtin_amdgcn_wmma_f32_16x16x32_bf16(false, a, false, b, (short)0, c,
                                                 false, false);
}

// A fragment (16x32 bf16, ISA 7.12.2): lane L holds row M=L&15,
// K = kb..kb+7 and 16+kb..16+kb+7 where kb = (L>=16)*8.
__device__ __forceinline__ bf16x16 load_a_frag(const __bf16* A, int lda, int kt,
                                               int m, int kb) {
  const __bf16* row = A + m * lda + kt * 32;
  bf16x8 lo = *(const bf16x8*)(row + kb);
  bf16x8 hi = *(const bf16x8*)(row + 16 + kb);
  return cat8(lo, hi);
}

// B fragment (32x16 bf16): lane L holds column N = nt*16 + (L&15),
// K = kb..kb+7 and 16+kb..16+kb+7 where kb = (L>=16)*8.
__device__ __forceinline__ bf16x16 load_b_frag(const __bf16* B, int ldb, int kt,
                                               int nt, int lane) {
  const __bf16* p = B + (kt * 32) * ldb + nt * 16 + (lane & 15);
  const int kb = (lane >> 4) * 8;
  bf16x16 v;
#pragma unroll
  for (int s = 0; s < 8; ++s) {
    v[s] = p[(kb + s) * ldb];
    v[8 + s] = p[(16 + kb + s) * ldb];
  }
  return v;
}

// ---------- Tensor Data Mover (CDNA5 TDM) ----------
#if defined(__HIP_DEVICE_COMPILE__) && defined(__gfx1250__) && \
    __has_builtin(__builtin_amdgcn_tensor_load_to_lds) &&      \
    __has_builtin(__builtin_amdgcn_s_wait_tensorcnt)
#define USE_TDM 1
#else
#define USE_TDM 0
#endif

#if USE_TDM
typedef unsigned int u32x4 __attribute__((ext_vector_type(4)));
typedef int i32x8 __attribute__((ext_vector_type(8)));
typedef int i32x4 __attribute__((ext_vector_type(4)));

// DMA a [tile1 x tile0] bf16 tile (row stride = stride0 elems) from global to
// LDS byte offset lds_off. Rows past dim1 / cols past dim0 read as zero (ISA
// Ch8 OOB rule) -- used for the N-row tail. D# fields per CDNA5 ISA Sec 8.
// This toolchain's builtin takes 6 args (g0, g1, g2, g3, extra, cpol).
__device__ __forceinline__ void tdm_load_bf16(unsigned lds_off, const void* gptr,
                                              unsigned dim0, unsigned dim1,
                                              unsigned tile0, unsigned tile1,
                                              unsigned stride0) {
  unsigned long long ga = (unsigned long long)gptr;
  u32x4 g0;
  g0.x = 1u;                                               // count=1, no gather
  g0.y = lds_off;                                          // lds_addr
  g0.z = (unsigned)ga;                                     // global_addr lo
  g0.w = ((unsigned)(ga >> 32) & 0x01FFFFFFu) | 0x80000000u;  // addr hi | type=2
  i32x8 g1;
  g1[0] = 0x00010000;                                      // data_size=1 (2B)
  g1[1] = (int)((dim0 & 0xFFFFu) << 16);                   // tensor_dim0 lo
  g1[2] = (int)(((dim0 >> 16) & 0xFFFFu) | ((dim1 & 0xFFFFu) << 16));
  g1[3] = (int)(((dim1 >> 16) & 0xFFFFu) | ((tile0 & 0xFFFFu) << 16));
  g1[4] = (int)(tile1 & 0xFFFFu);                          // tile_dim1, tile_dim2=0
  g1[5] = (int)stride0;                                    // tensor_dim0_stride lo
  g1[6] = 0;
  g1[7] = 0;
  i32x4 z4 = {0, 0, 0, 0};
  i32x8 z8 = {0, 0, 0, 0, 0, 0, 0, 0};
  __builtin_amdgcn_tensor_load_to_lds(g0, g1, z4, z4, z8, 0);
}
#endif

// ---------- small utility kernels ----------
__global__ void k_fill(float* p, float v, int n) {
  int i = blockIdx.x * blockDim.x + threadIdx.x;
  if (i < n) p[i] = v;
}

__global__ void k_cvt_bf16(const float* in, __bf16* out, int n) {
  int i = blockIdx.x * blockDim.x + threadIdx.x;
  if (i < n) out[i] = to_bf16(in[i]);
}

// Wc2 [128,10] f32 -> [128,16] bf16 zero-padded
__global__ void k_wc2_pad(const float* w, __bf16* out) {
  int i = blockIdx.x * blockDim.x + threadIdx.x;
  if (i < 128 * 16) {
    int k = i >> 4, c = i & 15;
    out[i] = to_bf16(c < 10 ? w[k * 10 + c] : 0.f);
  }
}

__global__ void k_deg(const int* dst, float* deg, int E) {
  int i = blockIdx.x * blockDim.x + threadIdx.x;
  if (i < E) atomicAdd(&deg[dst[i]], 1.0f);
}

__global__ void k_dinv(const float* deg, float* dinv, int n) {
  int i = blockIdx.x * blockDim.x + threadIdx.x;
  if (i < n) {
    float d = deg[i];
    dinv[i] = d > 0.f ? rsqrtf(d) : 0.f;
  }
}

// ---------- node-feature GEMM: out[N,128] = A[N,128] @ W[128,128] ----------
// 256 threads (8 waves), 128x128 output per block. A and W already bf16 in
// global; both tiles staged to LDS by the TDM (overlapping OOB-zero fill).
__global__ void k_gemm_node(const __bf16* A, const __bf16* Bw,
                            float* out, int nrows) {
  extern __shared__ __bf16 sm[];          // dynamic LDS: base offset 0
  __bf16* Al = sm;                        // 128*128 bf16 @ byte 0
  __bf16* Bl = sm + 128 * 128;            // 128*128 bf16 @ byte 32768
  const int tid = threadIdx.x;
  const int rowbase = blockIdx.x * 128;

#if USE_TDM
  if (tid < 32) {   // one wave issues both DMAs (EXEC ignored by TDM)
    tdm_load_bf16(0u, A + (size_t)rowbase * 128,
                  128u, (unsigned)(nrows - rowbase), 128u, 128u, 128u);
    tdm_load_bf16(32768u, Bw, 16384u, 1u, 16384u, 1u, 16384u);
  }
  __builtin_amdgcn_s_wait_tensorcnt(0);
  __syncthreads();
#else
  for (int i = tid; i < 128 * 128; i += 256) {
    int r = i >> 7, c = i & 127;
    int gr = rowbase + r;
    Al[i] = (gr < nrows) ? A[(size_t)gr * 128 + c] : to_bf16(0.f);
    Bl[i] = Bw[i];
  }
  __syncthreads();
#endif

  const int wave = tid >> 5, lane = tid & 31;
  const int m  = lane & 15;
  const int kb = (lane >> 4) * 8;
  const int n  = lane & 15;
  const int mb = (lane >> 4) * 8;
  const __bf16* Aw = &Al[(wave * 16) * 128];
  const bool full = (rowbase + 128) <= nrows;   // uniform per block

#pragma unroll
  for (int nt = 0; nt < 8; ++nt) {
    f32x8 acc;
#pragma unroll
    for (int r = 0; r < 8; ++r) acc[r] = 0.f;
#pragma unroll
    for (int kt = 0; kt < 4; ++kt) {
      bf16x16 a = load_a_frag(Aw, 128, kt, m, kb);
      bf16x16 b = load_b_frag(Bl, 128, kt, nt, lane);
      acc = wmma_bf16(a, b, acc);
    }
    const int row0 = rowbase + wave * 16 + mb;
    if (full) {
#pragma unroll
      for (int r = 0; r < 8; ++r)
        out[(row0 + r) * 128 + nt * 16 + n] = acc[r];
    } else {
#pragma unroll
      for (int r = 0; r < 8; ++r)
        if (row0 + r < nrows) out[(row0 + r) * 128 + nt * 16 + n] = acc[r];
    }
  }
}

// ---------- edge scatter: agg[dst] += h[src] * dinv[src]*dinv[dst] ----------
// one wave per edge, one float4 per lane (32*4 = 128 features)
__global__ void k_scatter(const float* h, const int* src, const int* dst,
                          const float* dinv, float* agg, int E) {
  int t = blockIdx.x * blockDim.x + threadIdx.x;
  int e = t >> 5;
  int lane = t & 31;
  if (e >= E) return;
  int s = src[e], d = dst[e];
  float norm = dinv[s] * dinv[d];
  const float4 v = *(const float4*)(h + (size_t)s * 128 + lane * 4);
  float* o = agg + (size_t)d * 128 + lane * 4;
  atomicAdd(o + 0, v.x * norm);
  atomicAdd(o + 1, v.y * norm);
  atomicAdd(o + 2, v.z * norm);
  atomicAdd(o + 3, v.w * norm);
}

// ---------- epilogue: relu(agg + dinv^2 * h (self-loop) + b) -> bf16 ----------
__global__ void k_bias_relu(const float* agg, const float* h, const float* dinv,
                            const float* b, __bf16* outb, int total) {
  int i = blockIdx.x * blockDim.x + threadIdx.x;
  if (i >= total) return;
  int f = i & 127;
  int node = i >> 7;
  float di = dinv[node];
  float v = agg[i] + di * di * h[i] + b[f];
  outb[i] = to_bf16(fmaxf(v, 0.f));
}

// ---------- fused edge MLP ----------
// out[e,0:10] = relu(concat(h2b[src],h2b[dst]) @ Wc1 + bc1) @ Wc2 + bc2
// 256 threads = 8 waves, one 16-edge tile per wave. Weights DMA'd to LDS by
// the TDM, overlapped with the per-edge feature gather. ~164 KB of the 320 KB
// WGP LDS -> one block per WGP.
__global__ void k_edge_mlp(const __bf16* h2b, const int* src, const int* dst,
                           const __bf16* wc1b, const float* bc1,
                           const __bf16* wc2b, const float* bc2,
                           float* out, int E) {
  extern __shared__ __bf16 sm[];        // dynamic LDS, base byte offset 0
  __bf16* W1s = sm;                     // 256*128 = 32768 elems @ byte 0
  __bf16* W2s = W1s + 256 * 128;        // 128*16  = 2048  elems @ byte 65536
  __bf16* As  = W2s + 128 * 16;         // 8*16*256 elems        @ byte 69632
  __bf16* Zs  = As + 8 * 16 * 256;      // 8*16*128 elems        @ byte 135168

  const int tid = threadIdx.x;
#if USE_TDM
  if (tid < 32) {
    tdm_load_bf16(0u,     wc1b, 32768u, 1u, 32768u, 1u, 32768u);
    tdm_load_bf16(65536u, wc2b, 2048u,  1u, 2048u,  1u, 2048u);
  }
#else
  for (int i = tid; i < 256 * 128; i += 256) W1s[i] = wc1b[i];
  for (int i = tid; i < 128 * 16; i += 256)  W2s[i] = wc2b[i];
#endif

  const int wave = tid >> 5, lane = tid & 31;
  const int ebase = (blockIdx.x * 8 + wave) * 16;
  __bf16* Aw = As + wave * 16 * 256;
  __bf16* Zw = Zs + wave * 16 * 128;
  const bool full = (ebase + 16) <= E;   // uniform per wave

  // gather (overlaps the TDM weight DMA): lanes 0..15 src feats, 16..31 dst
  if (full) {
#pragma unroll 4
    for (int r = 0; r < 16; ++r) {
      int idx = (lane < 16) ? src[ebase + r] : dst[ebase + r];
      uint4 v = *(const uint4*)(h2b + (size_t)idx * 128 + (lane & 15) * 8);
      *(uint4*)(Aw + r * 256 + lane * 8) = v;
    }
  } else {
    for (int r = 0; r < 16; ++r) {
      uint4 v = make_uint4(0u, 0u, 0u, 0u);
      if (ebase + r < E) {
        int idx = (lane < 16) ? src[ebase + r] : dst[ebase + r];
        v = *(const uint4*)(h2b + (size_t)idx * 128 + (lane & 15) * 8);
      }
      *(uint4*)(Aw + r * 256 + lane * 8) = v;
    }
  }
#if USE_TDM
  __builtin_amdgcn_s_wait_tensorcnt(0);
#endif
  __syncthreads();

  const int m  = lane & 15;
  const int kb = (lane >> 4) * 8;
  const int n  = lane & 15;
  const int mb = (lane >> 4) * 8;

  // GEMM1: [16,256] @ [256,128] + bc1, relu -> Z (bf16)
#pragma unroll
  for (int nt = 0; nt < 8; ++nt) {
    float bias = bc1[nt * 16 + n];
    f32x8 acc;
#pragma unroll
    for (int r = 0; r < 8; ++r) acc[r] = bias;
#pragma unroll
    for (int kt = 0; kt < 8; ++kt) {
      bf16x16 a = load_a_frag(Aw, 256, kt, m, kb);
      bf16x16 b = load_b_frag(W1s, 128, kt, nt, lane);
      acc = wmma_bf16(a, b, acc);
    }
#pragma unroll
    for (int r = 0; r < 8; ++r)
      Zw[(mb + r) * 128 + nt * 16 + n] = to_bf16(fmaxf(acc[r], 0.f));
  }
  __syncthreads();

  // GEMM2: [16,128] @ [128,16(pad of 10)] + bc2 -> out
  {
    float bias = (n < 10) ? bc2[n] : 0.f;
    f32x8 acc;
#pragma unroll
    for (int r = 0; r < 8; ++r) acc[r] = bias;
#pragma unroll
    for (int kt = 0; kt < 4; ++kt) {
      bf16x16 a = load_a_frag(Zw, 128, kt, m, kb);
      bf16x16 b = load_b_frag(W2s, 16, kt, 0, lane);
      acc = wmma_bf16(a, b, acc);
    }
    if (n < 10) {
      if (full) {
#pragma unroll
        for (int r = 0; r < 8; ++r)
          out[(ebase + mb + r) * 10 + n] = acc[r];
      } else {
#pragma unroll
        for (int r = 0; r < 8; ++r)
          if (ebase + mb + r < E) out[(ebase + mb + r) * 10 + n] = acc[r];
      }
    }
  }
}

// ---------- host launcher ----------
extern "C" void kernel_launch(void* const* d_in, const int* in_sizes, int n_in,
                              void* d_out, int out_size, void* d_ws, size_t ws_size,
                              hipStream_t stream) {
  const float* x   = (const float*)d_in[0];
  const int*   ei  = (const int*)d_in[1];
  const float* W1  = (const float*)d_in[2];
  const float* b1  = (const float*)d_in[3];
  const float* W2  = (const float*)d_in[4];
  const float* b2  = (const float*)d_in[5];
  const float* Wc1 = (const float*)d_in[6];
  const float* bc1 = (const float*)d_in[7];
  const float* Wc2 = (const float*)d_in[8];
  const float* bc2 = (const float*)d_in[9];
  float* out = (float*)d_out;

  const int N = in_sizes[0] / 128;
  const int E = in_sizes[1] / 2;
  const int NF = N * 128;
  const int* src = ei;
  const int* dst = ei + E;

  // workspace carve-up (all offsets 16B aligned)
  char* ws = (char*)d_ws;
  float* deg  = (float*)ws;   ws += (size_t)N * 4;
  float* dinv = (float*)ws;   ws += (size_t)N * 4;
  float* hbuf = (float*)ws;   ws += (size_t)NF * 4;
  float* agg  = (float*)ws;   ws += (size_t)NF * 4;
  __bf16* xb   = (__bf16*)ws; ws += (size_t)NF * 2;
  __bf16* h1b  = (__bf16*)ws; ws += (size_t)NF * 2;
  __bf16* h2b  = (__bf16*)ws; ws += (size_t)NF * 2;
  __bf16* w1b  = (__bf16*)ws; ws += (size_t)128 * 128 * 2;
  __bf16* w2b  = (__bf16*)ws; ws += (size_t)128 * 128 * 2;
  __bf16* wc1b = (__bf16*)ws; ws += (size_t)256 * 128 * 2;
  __bf16* wc2b = (__bf16*)ws;

  const int T = 256;
  dim3 blk(T);
#define GRID(n) dim3((unsigned)(((n) + T - 1) / T))

  // one-time bf16 conversions (operands for the WMMA/TDM paths)
  k_cvt_bf16<<<GRID(NF), blk, 0, stream>>>(x, xb, NF);
  k_cvt_bf16<<<GRID(128 * 128), blk, 0, stream>>>(W1, w1b, 128 * 128);
  k_cvt_bf16<<<GRID(128 * 128), blk, 0, stream>>>(W2, w2b, 128 * 128);
  k_cvt_bf16<<<GRID(256 * 128), blk, 0, stream>>>(Wc1, wc1b, 256 * 128);
  k_wc2_pad<<<GRID(128 * 16), blk, 0, stream>>>(Wc2, wc2b);

  // degrees (self-loop contributes the +1 init)
  k_fill<<<GRID(N), blk, 0, stream>>>(deg, 1.0f, N);
  k_deg<<<GRID(E), blk, 0, stream>>>(dst, deg, E);
  k_dinv<<<GRID(N), blk, 0, stream>>>(deg, dinv, N);

  const int gemm_blocks = (N + 127) / 128;
  const size_t gemm_lds = 2u * 128 * 128 * sizeof(__bf16);   // 64 KB
  const int scat_blocks = (int)(((long long)E * 32 + T - 1) / T);

  // ---- GCN layer 1 ----
  k_fill<<<GRID(NF), blk, 0, stream>>>(agg, 0.f, NF);
  k_gemm_node<<<gemm_blocks, blk, gemm_lds, stream>>>(xb, w1b, hbuf, N);
  k_scatter<<<scat_blocks, blk, 0, stream>>>(hbuf, src, dst, dinv, agg, E);
  k_bias_relu<<<GRID(NF), blk, 0, stream>>>(agg, hbuf, dinv, b1, h1b, NF);

  // ---- GCN layer 2 ----
  k_fill<<<GRID(NF), blk, 0, stream>>>(agg, 0.f, NF);
  k_gemm_node<<<gemm_blocks, blk, gemm_lds, stream>>>(h1b, w2b, hbuf, N);
  k_scatter<<<scat_blocks, blk, 0, stream>>>(hbuf, src, dst, dinv, agg, E);
  k_bias_relu<<<GRID(NF), blk, 0, stream>>>(agg, hbuf, dinv, b2, h2b, NF);

  // ---- fused edge MLP ----
  const size_t mlp_lds = (size_t)(256 * 128 + 128 * 16 + 8 * 16 * 256 +
                                  8 * 16 * 128) * sizeof(__bf16);  // ~164 KB
  int eblocks = (E + 127) / 128;
  k_edge_mlp<<<eblocks, blk, mlp_lds, stream>>>(h2b, src, dst, wc1b, bc1,
                                                wc2b, bc2, out, E);
#undef GRID
}
